// NeighborhoodModel_69990787055940
// MI455X (gfx1250) — matrix operand order repaired and checked by reading, not compiled, not run on hardware
//
#include <hip/hip_runtime.h>

// ---------------------------------------------------------------------------
// NeighborhoodModel forward for MI455X (gfx1250, wave32, WMMA + TDM).
// B=2, N=2048, E=64, H=2 heads, K=2 neighborhoods, 2 GNN layers.
// Dominant GEMM (alpha @ z, 16x [2048x2048x64]) runs on v_wmma_f32_16x16x32_bf16
// with softmax normalization deferred past the matmul (linear per row).
// Mask tiles are staged LDS-side by the Tensor Data Mover (tensor_load_to_lds).
// MCF scan collapsed to 9 transposed matvecs; dual scan is per-edge scalar.
// ---------------------------------------------------------------------------

#define BIGV 1000000000.0f
constexpr int Bc = 2, Nn = 2048, FEMB = 16, FFEAT = 2, FIN = 18, Ee = 64, Hh = 2, Kk = 2;
constexpr int GRAPH_LAYERS = 2, FLOW_ITERS = 10, DUAL_ITERS = 10;
constexpr float STEPc = 0.1f, MOMc = 0.9f;

typedef __attribute__((ext_vector_type(16))) unsigned short v16u;
typedef __attribute__((ext_vector_type(16))) __bf16         v16bf;
typedef __attribute__((ext_vector_type(8)))  float          v8f;
typedef __attribute__((ext_vector_type(4)))  unsigned int   v4u;
typedef __attribute__((ext_vector_type(8)))  int            v8i;
typedef __attribute__((ext_vector_type(4)))  int            v4i;

__device__ __forceinline__ unsigned short f2bf(float f) {
  unsigned u = __float_as_uint(f);
  u += 0x7fffu + ((u >> 16) & 1u);          // round-to-nearest-even
  return (unsigned short)(u >> 16);
}

// --------------------------- encoder: h = tanh(tanh(xW0+b0)W1+b1) -----------
__global__ void k_encoder(const float* __restrict__ emb, const float* __restrict__ feat,
                          const float* __restrict__ W0, const float* __restrict__ b0,
                          const float* __restrict__ W1, const float* __restrict__ b1,
                          float* __restrict__ h) {
  const int n = blockIdx.x, b = blockIdx.y, e = threadIdx.x;
  __shared__ float x[FIN];
  __shared__ float t0[Ee];
  const size_t row = (size_t)b * Nn + n;
  if (e < FEMB) x[e] = emb[row * FEMB + e];
  else if (e < FIN) x[e] = feat[row * FFEAT + (e - FEMB)];
  __syncthreads();
  float a = b0[e];
  #pragma unroll
  for (int d = 0; d < FIN; ++d) a = fmaf(x[d], W0[d * Ee + e], a);
  t0[e] = tanhf(a);
  __syncthreads();
  float a1 = b1[e];
  for (int d = 0; d < Ee; ++d) a1 = fmaf(t0[d], W1[d * Ee + e], a1);
  h[row * Ee + e] = tanhf(a1);
}

// ---------- per-head projection: z = h@attn_W[h]; s_src/s_dst; zT in bf16 ---
__global__ void k_proj(const float* __restrict__ h, const float* __restrict__ attnW,
                       const float* __restrict__ asrc, const float* __restrict__ adst,
                       unsigned short* __restrict__ zT, float* __restrict__ ssrc,
                       float* __restrict__ sdst) {
  const int n = blockIdx.x, hb = blockIdx.y;
  const int hd = hb >> 1, b = hb & 1, e = threadIdx.x;
  __shared__ float hr[Ee], zr[Ee];
  const size_t row = (size_t)b * Nn + n;
  hr[e] = h[row * Ee + e];
  __syncthreads();
  const float* W = attnW + (size_t)hd * Ee * Ee;
  float a = 0.f;
  for (int d = 0; d < Ee; ++d) a = fmaf(hr[d], W[d * Ee + e], a);
  zr[e] = a;
  zT[(((size_t)hd * Bc + b) * Ee + e) * Nn + n] = f2bf(a);   // z transposed [h][b][e][m]
  __syncthreads();
  if (e == 0) {
    float ss = 0.f, sd = 0.f;
    for (int d = 0; d < Ee; ++d) {
      ss = fmaf(zr[d], asrc[hd * Ee + d], ss);
      sd = fmaf(zr[d], adst[hd * Ee + d], sd);
    }
    const size_t idx = ((size_t)hd * Bc + b) * Nn + n;
    ssrc[idx] = ss; sdst[idx] = sd;
  }
}

// ---------------- attention aggregation: agg = softmax(mask) @ z  -----------
// grid.x = N/16 row-block, grid.y = ((h*B)+b)*K + k.  256 threads / 8 waves.
// TDM stages each 16xCH f32 mask tile into LDS; alpha tile built as bf16
// exp(score - rowmax) (unnormalized: 1/rowsum folded into the accumulator),
// consumed by v_wmma_f32_16x16x32_bf16.
__global__ void __launch_bounds__(256, 2) k_attn(
    const float* __restrict__ nbrs, const float* __restrict__ ssrc,
    const float* __restrict__ sdst, const unsigned short* __restrict__ zT,
    float* __restrict__ agg_raw) {
  constexpr int CH = 512;
  __shared__ float          maskS[16][CH];          // 32 KB, TDM-filled
  __shared__ unsigned short alphaS[16][CH + 8];     // 16.6 KB
  __shared__ float red[16][16];
  __shared__ float rowmaxS[16], rowinv[16];
  __shared__ float pacc[4][8][32];

  const int rb = blockIdx.x;
  const int y  = blockIdx.y;
  const int k  = y & 1, b = (y >> 1) & 1, hd = y >> 2;

  const int tid = threadIdx.x;
  const int r = tid >> 4, seg = tid & 15;           // fill roles: row / col lane
  const int rowg = rb * 16 + r;

  const float* tile0 = nbrs + (((size_t)k * Bc + b) * Nn + (size_t)rb * 16) * Nn;
  const float* nbrp  = tile0 + (size_t)r * Nn;
  const float* sd    = sdst + ((size_t)hd * Bc + b) * Nn;
  const float  ssr   = ssrc[((size_t)hd * Bc + b) * Nn + rowg];

  // ---- pass 1: masked max of s_dst only (tanh is monotonic -> one tanh/row)
  float lmax = -INFINITY;
  for (int q = 0; q < Nn / 16; ++q) {
    const int c = q * 16 + seg;
    if (nbrp[c] > 0.f) lmax = fmaxf(lmax, sd[c]);
  }
  red[r][seg] = lmax;
  __syncthreads();
  if (seg == 0) {
    float m = red[r][0];
    for (int q = 1; q < 16; ++q) m = fmaxf(m, red[r][q]);
    rowmaxS[r] = (m > -INFINITY) ? tanhf(ssr + m) : -BIGV;  // empty row -> -BIG
  }
  __syncthreads();
  const float rm = rowmaxS[r];

  // ---- TDM descriptor (block-uniform): 2D tile 16 x CH f32, row stride N --
  v4u g0;
  g0[0] = 1u;                                   // count = 1 (valid user D#)
  g0[1] = (unsigned)(size_t)(&maskS[0][0]);     // lds_addr (byte offset)
  g0[2] = 0u; g0[3] = 0u;                       // global_addr set per chunk
  v8i g1;
  g1[0] = (int)(2u << 16);                      // data_size = 4 bytes
  g1[1] = (int)((unsigned)CH << 16);            // tensor_dim0 = CH
  g1[2] = (int)(16u << 16);                     // tensor_dim1 = 16
  g1[3] = (int)((unsigned)CH << 16);            // tile_dim0 = CH
  g1[4] = 16;                                   // tile_dim1 = 16
  g1[5] = Nn;                                   // tensor_dim0_stride = N
  g1[6] = 0;
  g1[7] = 0;
  const v4i gz = {0, 0, 0, 0};
  const v8i gz8 = {0, 0, 0, 0, 0, 0, 0, 0};

  const int wave = tid >> 5, lane = tid & 31;
  const int ct = wave & 3, mh = wave >> 2;          // col tile / m-half in chunk
  const int arow = lane & 15, akoff = (lane >> 4) * 16;
  const int bcol = lane & 15, bkoff = (lane >> 4) * 16;
  const unsigned short* zb =
      zT + (((size_t)hd * Bc + b) * Ee + (ct * 16 + bcol)) * Nn;

  v8f acc = {0.f, 0.f, 0.f, 0.f, 0.f, 0.f, 0.f, 0.f};
  float lsum = 0.f;

  for (int cs = 0; cs < Nn; cs += CH) {
    if (wave == 0) {                               // one DMA per block chunk
      const unsigned long long ga = (unsigned long long)(size_t)(tile0 + cs);
      g0[2] = (unsigned)ga;
      g0[3] = ((unsigned)(ga >> 32) & 0x01ffffffu) | (2u << 30);  // addr|type=2
      __builtin_amdgcn_tensor_load_to_lds(g0, g1, gz, gz, gz8, 0);
      __builtin_amdgcn_s_wait_tensorcnt(0);
    }
    __syncthreads();                               // publish TDM-written maskS

    // fill alpha tile (bf16 exp) + accumulate row sum in the same pass
    for (int q = 0; q < CH / 16; ++q) {
      const int cl = q * 16 + seg;
      const float v = maskS[r][cl] > 0.f ? tanhf(ssr + sd[cs + cl]) : -BIGV;
      const float ex = __expf(v - rm);
      lsum += ex;
      alphaS[r][cl] = f2bf(ex);
    }
    __syncthreads();

    for (int m0 = mh * (CH / 2); m0 < (mh + 1) * (CH / 2); m0 += 32) {
      v16u au, bu;
      #pragma unroll
      for (int j = 0; j < 8; ++j) {
        au[j]     = alphaS[arow][m0 + akoff + j];
        au[8 + j] = alphaS[arow][m0 + akoff + 8 + j];
      }
      #pragma unroll
      for (int j = 0; j < 16; ++j) bu[j] = zb[cs + m0 + bkoff + j];
      acc = __builtin_amdgcn_wmma_f32_16x16x32_bf16(
          false, __builtin_bit_cast(v16bf, au),
          false, __builtin_bit_cast(v16bf, bu),
          (short)0, acc, false, false);
    }
    __syncthreads();                               // LDS tiles reused next chunk
  }

  // ---- row-sum reduce, pairwise wave reduce (m-halves), scale, store ------
  red[r][seg] = lsum;
  if (wave >= 4) {
    #pragma unroll
    for (int v = 0; v < 8; ++v) pacc[ct][v][lane] = acc[v];
  }
  __syncthreads();
  if (seg == 0) {
    float s = 0.f;
    for (int q = 0; q < 16; ++q) s += red[r][q];
    rowinv[r] = 1.f / s;                           // deferred normalization
  }
  __syncthreads();
  if (wave < 4) {
    float* outp = agg_raw + ((((size_t)hd * Bc + b) * Kk + k) * Nn) * Ee;
    #pragma unroll
    for (int v = 0; v < 8; ++v) {
      const int M = (lane >> 4) * 8 + v;
      const float val = (acc[v] + pacc[ct][v][lane]) * rowinv[M];
      outp[(size_t)(rb * 16 + M) * Ee + (ct * 16 + (lane & 15))] = val;
    }
  }
}

// ------------- neighborhood mix + GRU cell, h updated in place --------------
__global__ void k_gru(float* __restrict__ h, const float* __restrict__ agg_raw,
                      const float* __restrict__ nbrq, const float* __restrict__ gW,
                      const float* __restrict__ gU, const float* __restrict__ gb) {
  const int n = blockIdx.x, b = blockIdx.y, e = threadIdx.x;
  __shared__ float a0[Ee], a1[Ee], hr[Ee], nxt[Ee], rh[Ee], cf[2];
  const size_t row = (size_t)b * Nn + n;
  const size_t b0 = ((((size_t)0 * Bc + b) * Kk + 0) * Nn + n) * Ee + e;  // h=0,k=0
  const size_t b1 = ((((size_t)1 * Bc + b) * Kk + 0) * Nn + n) * Ee + e;  // h=1,k=0
  a0[e] = tanhf(0.5f * (agg_raw[b0] + agg_raw[b1]));
  a1[e] = tanhf(0.5f * (agg_raw[b0 + (size_t)Nn * Ee] + agg_raw[b1 + (size_t)Nn * Ee]));
  hr[e] = h[row * Ee + e];
  __syncthreads();
  if (e == 0) {
    float s0 = 0.f, s1 = 0.f;
    for (int d = 0; d < Ee; ++d) { s0 = fmaf(a0[d], nbrq[d], s0); s1 = fmaf(a1[d], nbrq[d], s1); }
    s0 = tanhf(s0); s1 = tanhf(s1);
    const float mx = fmaxf(s0, s1);
    const float e0 = __expf(s0 - mx), e1 = __expf(s1 - mx);
    const float inv = 1.f / (e0 + e1);
    cf[0] = e0 * inv; cf[1] = e1 * inv;
  }
  __syncthreads();
  nxt[e] = cf[0] * a0[e] + cf[1] * a1[e];
  __syncthreads();
  float az = gb[e], ar = gb[Ee + e];
  for (int d = 0; d < Ee; ++d) {
    az = fmaf(nxt[d], gW[d * 3 * Ee + e], az);
    az = fmaf(hr[d],  gU[d * 3 * Ee + e], az);
    ar = fmaf(nxt[d], gW[d * 3 * Ee + Ee + e], ar);
    ar = fmaf(hr[d],  gU[d * 3 * Ee + Ee + e], ar);
  }
  const float zg = 1.f / (1.f + __expf(-az));
  const float rr = 1.f / (1.f + __expf(-ar));
  rh[e] = rr * hr[e];
  __syncthreads();
  float ah = gb[2 * Ee + e];
  for (int d = 0; d < Ee; ++d) {
    ah = fmaf(nxt[d], gW[d * 3 * Ee + 2 * Ee + e], ah);
    ah = fmaf(rh[d],  gU[d * 3 * Ee + 2 * Ee + e], ah);
  }
  const float ht = tanhf(ah);
  h[row * Ee + e] = (1.f - zg) * hr[e] + zg * ht;
}

// ------------- decoder + dual heads; dual_demand accumulated ---------------
__global__ void k_heads(const float* __restrict__ h,
                        const float* dW0, const float* db0, const float* dW1, const float* db1,
                        const float* uW0, const float* ub0, const float* uW1, const float* ub1,
                        const float* __restrict__ dem,
                        float* __restrict__ nw, float* __restrict__ dv,
                        float* __restrict__ dd_acc) {
  const int n = blockIdx.x, b = blockIdx.y, e = threadIdx.x;
  __shared__ float hr[Ee], t0[Ee], t1[Ee];
  const size_t row = (size_t)b * Nn + n;
  hr[e] = h[row * Ee + e];
  __syncthreads();
  float a0 = db0[e], a1 = ub0[e];
  for (int d = 0; d < Ee; ++d) { a0 = fmaf(hr[d], dW0[d * Ee + e], a0); a1 = fmaf(hr[d], uW0[d * Ee + e], a1); }
  t0[e] = tanhf(a0); t1[e] = tanhf(a1);
  __syncthreads();
  if (e == 0) {
    float s0 = db1[0], s1 = ub1[0];
    for (int d = 0; d < Ee; ++d) { s0 = fmaf(t0[d], dW1[d], s0); s1 = fmaf(t1[d], uW1[d], s1); }
    nw[row] = s0; dv[row] = s1;
    atomicAdd(&dd_acc[b], s1 * dem[row]);
  }
}

// ------------- prop = row-softmax(adj ? nw[j] : -BIG), materialized --------
__global__ void k_prop(const float* __restrict__ adj, const float* __restrict__ nw,
                       float* __restrict__ prop) {
  const int i = blockIdx.x, b = blockIdx.y, t = threadIdx.x;
  __shared__ float red[256];
  const float* arow = adj + ((size_t)b * Nn + i) * Nn;
  const float* nwb  = nw + (size_t)b * Nn;
  float lmax = -INFINITY;
  #pragma unroll
  for (int q = 0; q < 8; ++q) {
    const int j = t + q * 256;
    lmax = fmaxf(lmax, arow[j] > 0.f ? nwb[j] : -BIGV);
  }
  red[t] = lmax; __syncthreads();
  for (int s = 128; s > 0; s >>= 1) { if (t < s) red[t] = fmaxf(red[t], red[t + s]); __syncthreads(); }
  const float rmax = red[0];
  __syncthreads();
  float ev[8], lsum = 0.f;
  #pragma unroll
  for (int q = 0; q < 8; ++q) {
    const int j = t + q * 256;
    const float w = arow[j] > 0.f ? nwb[j] : -BIGV;
    ev[q] = __expf(w - rmax);
    lsum += ev[q];
  }
  red[t] = lsum; __syncthreads();
  for (int s = 128; s > 0; s >>= 1) { if (t < s) red[t] += red[t + s]; __syncthreads(); }
  const float inv = 1.f / red[0];
  float* prow = prop + ((size_t)b * Nn + i) * Nn;
  #pragma unroll
  for (int q = 0; q < 8; ++q) prow[t + q * 256] = ev[q] * inv;
}

// ------------- MCF: s1 = relu(-d); s <- relu(P^T s - d), 9 times -----------
__global__ void k_s_init(const float* __restrict__ dem, float* __restrict__ s) {
  const int idx = blockIdx.x * 256 + threadIdx.x;
  if (idx < Bc * Nn) s[idx] = fmaxf(-dem[idx], 0.f);
}

__global__ void k_mcf(const float* __restrict__ prop, const float* __restrict__ s_in,
                      const float* __restrict__ dem, float* __restrict__ s_out) {
  const int j = blockIdx.x * 256 + threadIdx.x;
  const int b = blockIdx.y;
  const float* P  = prop + (size_t)b * Nn * Nn;
  const float* si = s_in + (size_t)b * Nn;
  float acc = 0.f;
  for (int i = 0; i < Nn; ++i) {
    if (i + 8 < Nn) __builtin_prefetch(&P[(size_t)(i + 8) * Nn + j], 0, 0);
    acc = fmaf(P[(size_t)i * Nn + j], si[i], acc);
  }
  s_out[(size_t)b * Nn + j] = fmaxf(acc - dem[(size_t)b * Nn + j], 0.f);
}

// ------------- flow cost: sum (f_ij - adj*min(f_ij, f_ji))^2 ---------------
__global__ void k_flowcost(const float* __restrict__ prop, const float* __restrict__ adj,
                           const float* __restrict__ s, float* __restrict__ fc) {
  const int i = blockIdx.x, b = blockIdx.y, t = threadIdx.x;
  __shared__ float red[256];
  const float* P  = prop + (size_t)b * Nn * Nn;
  const float* A  = adj + (size_t)b * Nn * Nn;
  const float* sb = s + (size_t)b * Nn;
  const float si = sb[i];
  float acc = 0.f;
  for (int j = t; j < Nn; j += 256) {
    const float fij = P[(size_t)i * Nn + j] * si;
    const float fji = P[(size_t)j * Nn + i] * sb[j];
    const float v = fij - A[(size_t)i * Nn + j] * fminf(fij, fji);
    acc = fmaf(v, v, acc);
  }
  red[t] = acc; __syncthreads();
  for (int s2 = 128; s2 > 0; s2 >>= 1) { if (t < s2) red[t] += red[t + s2]; __syncthreads(); }
  if (t == 0) atomicAdd(&fc[b], red[0]);
}

// ------------- dual: per-edge 10-step scalar recursion ---------------------
__global__ void k_dual(const float* __restrict__ adj, const float* __restrict__ dv,
                       float* __restrict__ dc) {
  const int i = blockIdx.x, b = blockIdx.y, t = threadIdx.x;
  __shared__ float red[256];
  const float* A = adj + (size_t)b * Nn * Nn;
  const float* d = dv + (size_t)b * Nn;
  const float di = d[i];
  float acc = 0.f;
  for (int j = t; j < Nn; j += 256) {
    const float a = A[(size_t)i * Nn + j];
    const float D = a * (di - d[j]);
    float y = 0.f, m = 0.f;
    #pragma unroll
    for (int it = 0; it < DUAL_ITERS; ++it) {
      const float g = 2.f * y - D;
      m = MOMc * m + g;
      y = a * fmaxf(y - STEPc * m, 0.f);
    }
    acc += y * y - D * y;
  }
  red[t] = acc; __syncthreads();
  for (int s2 = 128; s2 > 0; s2 >>= 1) { if (t < s2) red[t] += red[t + s2]; __syncthreads(); }
  if (t == 0) atomicAdd(&dc[b], red[0]);
}

__global__ void k_zero(float* acc) { if (threadIdx.x < 8) acc[threadIdx.x] = 0.f; }

__global__ void k_final(const float* fc, const float* dc, const float* dd, float* out) {
  const int b = threadIdx.x;
  if (b < Bc) out[b] = fc[b] - (dc[b] - dd[b]);
}

// ---------------------------------------------------------------------------
extern "C" void kernel_launch(void* const* d_in, const int* in_sizes, int n_in,
                              void* d_out, int out_size, void* d_ws, size_t ws_size,
                              hipStream_t stream) {
  (void)in_sizes; (void)n_in; (void)out_size; (void)ws_size;
  const float* feat  = (const float*)d_in[0];
  const float* emb   = (const float*)d_in[1];
  const float* dem   = (const float*)d_in[2];
  const float* adj   = (const float*)d_in[3];
  const float* nbrs  = (const float*)d_in[4];
  const float* encW0 = (const float*)d_in[5];
  const float* encb0 = (const float*)d_in[6];
  const float* encW1 = (const float*)d_in[7];
  const float* encb1 = (const float*)d_in[8];
  const float* attnW = (const float*)d_in[9];
  const float* asrc  = (const float*)d_in[10];
  const float* adst  = (const float*)d_in[11];
  const float* nbrq  = (const float*)d_in[12];
  const float* gruW  = (const float*)d_in[13];
  const float* gruU  = (const float*)d_in[14];
  const float* grub  = (const float*)d_in[15];
  const float* decW0 = (const float*)d_in[16];
  const float* decb0 = (const float*)d_in[17];
  const float* decW1 = (const float*)d_in[18];
  const float* decb1 = (const float*)d_in[19];
  const float* duW0  = (const float*)d_in[20];
  const float* dub0  = (const float*)d_in[21];
  const float* duW1  = (const float*)d_in[22];
  const float* dub1  = (const float*)d_in[23];

  char* ws = (char*)d_ws;
  float*          h    = (float*)(ws + 0);                 // 1.0 MB
  unsigned short* zT   = (unsigned short*)(ws + 1048576);  // 1.0 MB (bf16, [h][b][e][m])
  float*          ssrc = (float*)(ws + 2097152);
  float*          sdst = (float*)(ws + 2129920);
  float*          agg  = (float*)(ws + 2162688);           // 8.0 MB [h][b][k][n][e]
  float*          nw   = (float*)(ws + 10551296);
  float*          dv   = (float*)(ws + 10567680);
  float*          prop = (float*)(ws + 10584064);          // 33.5 MB
  float*          sA   = (float*)(ws + 44138496);
  float*          sB   = (float*)(ws + 44154880);
  float*          accb = (float*)(ws + 44171264);          // fc[2] dc[2] dd[2]

  k_zero<<<1, 64, 0, stream>>>(accb);
  k_encoder<<<dim3(Nn, Bc), Ee, 0, stream>>>(emb, feat, encW0, encb0, encW1, encb1, h);

  for (int L = 0; L < GRAPH_LAYERS; ++L) {
    k_proj<<<dim3(Nn, Hh * Bc), Ee, 0, stream>>>(h, attnW, asrc, adst, zT, ssrc, sdst);
    k_attn<<<dim3(Nn / 16, Hh * Bc * Kk), 256, 0, stream>>>(nbrs, ssrc, sdst, zT, agg);
    k_gru<<<dim3(Nn, Bc), Ee, 0, stream>>>(h, agg, nbrq, gruW, gruU, grub);
  }

  k_heads<<<dim3(Nn, Bc), Ee, 0, stream>>>(h, decW0, decb0, decW1, decb1,
                                           duW0, dub0, duW1, dub1, dem, nw, dv, accb + 4);
  k_prop<<<dim3(Nn, Bc), 256, 0, stream>>>(adj, nw, prop);

  k_s_init<<<(Bc * Nn) / 256, 256, 0, stream>>>(dem, sA);
  float* bufs[2] = {sA, sB};
  for (int it = 0; it < FLOW_ITERS - 1; ++it)
    k_mcf<<<dim3(Nn / 256, Bc), 256, 0, stream>>>(prop, bufs[it & 1], dem, bufs[(it + 1) & 1]);
  // FLOW_ITERS-1 = 9 matvecs -> final s lands in sB

  k_flowcost<<<dim3(Nn, Bc), 256, 0, stream>>>(prop, adj, sB, accb + 0);
  k_dual<<<dim3(Nn, Bc), 256, 0, stream>>>(adj, dv, accb + 2);
  k_final<<<1, 64, 0, stream>>>(accb + 0, accb + 2, accb + 4, (float*)d_out);
}
